// SimpleALIFRNN_52759378264213
// MI455X (gfx1250) — matrix operand order repaired and checked by reading, not compiled, not run on hardware
//
#include <hip/hip_runtime.h>
#include <stdint.h>

// ---- problem dims (fixed by reference) ----
#define T_STEPS 500
#define BATCH   256
#define IN_DIM  256
#define HID     1024
#define OUT_DIM 128
#define KTOT    (IN_DIM + HID)   // 1280

#define THETA0 0.01f
#define BETA   1.8f

typedef __attribute__((ext_vector_type(16))) __bf16 v16bf;
typedef __attribute__((ext_vector_type(8)))  float  v8f;

union FragBF {
  uint32_t u32[8];
  uint4    q[2];
  v16bf    v;
};
union FragF {
  float f[8];
  v8f   v;
};

__device__ __forceinline__ uint16_t f2bf(float f) {
  uint32_t x = __float_as_uint(f);
  x += 0x7FFFu + ((x >> 16) & 1u);          // round-to-nearest-even
  return (uint16_t)(x >> 16);
}
__device__ __forceinline__ float bf2f(uint16_t h) {
  return __uint_as_float(((uint32_t)h) << 16);
}
__device__ __forceinline__ uint32_t pack2(float a, float b) {
  return (uint32_t)f2bf(a) | ((uint32_t)f2bf(b) << 16);
}

__device__ __forceinline__ void load_a_bf16(FragBF& af, const uint16_t* src, int half8) {
  af.q[0] = *(const uint4*)(src + half8);
  af.q[1] = *(const uint4*)(src + 16 + half8);
}
__device__ __forceinline__ void load_a_f32cvt(FragBF& af, const float* src, int half8) {
  const float4* p0 = (const float4*)(src + half8);
  const float4* p1 = (const float4*)(src + 16 + half8);
  float4 c0 = p0[0], c1 = p0[1];
  float4 d0 = p1[0], d1 = p1[1];
  af.u32[0] = pack2(c0.x, c0.y); af.u32[1] = pack2(c0.z, c0.w);
  af.u32[2] = pack2(c1.x, c1.y); af.u32[3] = pack2(c1.z, c1.w);
  af.u32[4] = pack2(d0.x, d0.y); af.u32[5] = pack2(d0.z, d0.w);
  af.u32[6] = pack2(d1.x, d1.y); af.u32[7] = pack2(d1.z, d1.w);
}

// ---------------------------------------------------------------------------
// init: W_h / W_o -> bf16 col-major, decays, zero states, optional x -> bf16
// ---------------------------------------------------------------------------
__global__ void init_kernel(const float* __restrict__ x,
                            const float* __restrict__ Wh, const float* __restrict__ Wo,
                            const float* __restrict__ tau_mem, const float* __restrict__ tau_adp,
                            const float* __restrict__ tau_out,
                            uint16_t* __restrict__ WhT, uint16_t* __restrict__ WoT,
                            uint16_t* __restrict__ xb,      // may be null (ws too small)
                            uint16_t* __restrict__ zb0, uint16_t* __restrict__ zb1,
                            float* __restrict__ u, float* __restrict__ a, float* __restrict__ ou,
                            float* __restrict__ alpha, float* __restrict__ rho,
                            float* __restrict__ aout, int* __restrict__ ns)
{
  const int stride = gridDim.x * blockDim.x;
  const int tid0 = blockIdx.x * blockDim.x + threadIdx.x;

  for (int i = tid0; i < KTOT * HID; i += stride) {        // Wh [k][n] row-major
    int k = i / HID, n = i - k * HID;
    WhT[(size_t)n * KTOT + k] = f2bf(Wh[i]);
  }
  for (int i = tid0; i < HID * OUT_DIM; i += stride) {     // Wo [k][o] row-major
    int k = i / OUT_DIM, o = i - k * OUT_DIM;
    WoT[(size_t)o * HID + k] = f2bf(Wo[i]);
  }
  for (int i = tid0; i < BATCH * HID; i += stride) {
    zb0[i] = 0; zb1[i] = 0; u[i] = 0.0f; a[i] = 0.0f;
  }
  for (int i = tid0; i < BATCH * OUT_DIM; i += stride) ou[i] = 0.0f;
  for (int i = tid0; i < HID; i += stride) {
    alpha[i] = __expf(-1.0f / tau_mem[i]);
    rho[i]   = __expf(-1.0f / tau_adp[i]);
  }
  for (int i = tid0; i < OUT_DIM; i += stride) aout[i] = __expf(-1.0f / tau_out[i]);
  if (xb) {
    const size_t NX = (size_t)T_STEPS * BATCH * IN_DIM;
    for (size_t i = tid0; i < NX; i += (size_t)stride) xb[i] = f2bf(x[i]);
  }
  if (tid0 == 0) *ns = 0;
}

// ---------------------------------------------------------------------------
// step A: cur = [x_t, z_old] @ W_h (WMMA bf16), fused ALIF update + spike
// 32x64 register blocking per wave: 2 A frags x 4 B frags -> 8 accumulators.
// grid = 32 blocks x 128 threads (4 waves); 128 tiles of 32x64 over [B,H]
// ---------------------------------------------------------------------------
template <bool XB>
__global__ void __launch_bounds__(128)
step_rec_kernel(const float* __restrict__ x, const uint16_t* __restrict__ xb, int t,
                const uint16_t* __restrict__ WhT,
                const uint16_t* __restrict__ zb_old, uint16_t* __restrict__ zb_new,
                float* __restrict__ u, float* __restrict__ a,
                const float* __restrict__ alpha, const float* __restrict__ rho,
                int* __restrict__ ns)
{
  const int lane  = threadIdx.x & 31;
  const int wave  = threadIdx.x >> 5;
  const int tile  = blockIdx.x * 4 + wave;   // 0..127
  const int mt    = tile >> 4;               // 0..7   (B tiles of 32)
  const int nt    = tile & 15;               // 0..15  (H tiles of 64)
  const int half  = lane >> 4;
  const int half8 = half * 8;
  const int col   = lane & 15;

  const int rowA0 = mt * 32 + col;           // A rows for this lane (two 16-row tiles)
  const int rowA1 = rowA0 + 16;
  const size_t xoff = (size_t)t * BATCH * IN_DIM;
  const float*    xrow0  = x + xoff + (size_t)rowA0 * IN_DIM;
  const float*    xrow1  = x + xoff + (size_t)rowA1 * IN_DIM;
  const uint16_t* xbrow0 = XB ? (xb + xoff + (size_t)rowA0 * IN_DIM) : nullptr;
  const uint16_t* xbrow1 = XB ? (xb + xoff + (size_t)rowA1 * IN_DIM) : nullptr;
  const uint16_t* zrow0  = zb_old + (size_t)rowA0 * HID;
  const uint16_t* zrow1  = zb_old + (size_t)rowA1 * HID;
  const uint16_t* wcol   = WhT + (size_t)(nt * 64 + col) * KTOT;  // B col base
  const int koffB = half * 8;

  FragF acc[2][4];
#pragma unroll
  for (int m = 0; m < 2; ++m)
#pragma unroll
    for (int s = 0; s < 4; ++s)
#pragma unroll
      for (int i = 0; i < 8; ++i) acc[m][s].f[i] = 0.0f;

#pragma unroll 2
  for (int kb = 0; kb < KTOT; kb += 32) {
    FragBF af[2];
    if (XB) {
      if (kb < IN_DIM) {
        load_a_bf16(af[0], xbrow0 + kb, half8);
        load_a_bf16(af[1], xbrow1 + kb, half8);
      } else {
        load_a_bf16(af[0], zrow0 + (kb - IN_DIM), half8);
        load_a_bf16(af[1], zrow1 + (kb - IN_DIM), half8);
      }
    } else if (kb < IN_DIM) {
      load_a_f32cvt(af[0], xrow0 + kb, half8);
      load_a_f32cvt(af[1], xrow1 + kb, half8);
    } else {
      load_a_bf16(af[0], zrow0 + (kb - IN_DIM), half8);
      load_a_bf16(af[1], zrow1 + (kb - IN_DIM), half8);
    }
    FragBF bf[4];
#pragma unroll
    for (int s = 0; s < 4; ++s) {
      const uint16_t* wc = wcol + (size_t)s * 16 * KTOT + kb + koffB;
      bf[s].q[0] = *(const uint4*)(wc);
      bf[s].q[1] = *(const uint4*)(wc + 16);
    }
#pragma unroll
    for (int s = 0; s < 4; ++s) {
#pragma unroll
      for (int m = 0; m < 2; ++m) {
        acc[m][s].v = __builtin_amdgcn_wmma_f32_16x16x32_bf16(
            false, af[m].v, false, bf[s].v, (short)0, acc[m][s].v, false, false);
      }
    }
  }

  // fused ALIF epilogue: lane owns (M = m*16 + half*8 + r, N = s*16 + col)
  int sp = 0;
#pragma unroll
  for (int s = 0; s < 4; ++s) {
    const int h = nt * 64 + s * 16 + col;
    const float al = alpha[h], rh = rho[h];
    const float oma = 1.0f - al, omr = 1.0f - rh;
#pragma unroll
    for (int m = 0; m < 2; ++m) {
#pragma unroll
      for (int r = 0; r < 8; ++r) {
        const int b = mt * 32 + m * 16 + half8 + r;
        const size_t idx = (size_t)b * HID + h;
        const float zo = bf2f(zb_old[idx]);
        const float an = rh * a[idx] + omr * zo;
        const float theta = THETA0 + BETA * an;
        const float un = al * u[idx] + oma * acc[m][s].f[r] - zo * theta;
        const int fire = (un - theta) > 0.0f ? 1 : 0;
        a[idx] = an;
        u[idx] = un;
        zb_new[idx] = fire ? (uint16_t)0x3F80u : (uint16_t)0u;  // bf16 1.0 / 0.0
        sp += fire;
      }
    }
  }
  for (int off = 16; off > 0; off >>= 1) sp += __shfl_down(sp, off, 32);
  if (lane == 0) atomicAdd(ns, sp);
}

// ---------------------------------------------------------------------------
// step C: z_new @ W_o (WMMA bf16), fused leaky-integrator readout
// 32x64 blocking; grid = 4 blocks x 128 threads; 16 tiles over [B,O]
// ---------------------------------------------------------------------------
__global__ void __launch_bounds__(128)
step_out_kernel(const uint16_t* __restrict__ zb_new,
                const uint16_t* __restrict__ WoT,
                float* __restrict__ ou, const float* __restrict__ aout,
                float* __restrict__ outputs, int t)
{
  const int lane  = threadIdx.x & 31;
  const int wave  = threadIdx.x >> 5;
  const int tile  = blockIdx.x * 4 + wave;   // 0..15
  const int mt    = tile >> 1;               // 0..7
  const int nt    = tile & 1;                // 0..1
  const int half  = lane >> 4;
  const int half8 = half * 8;
  const int col   = lane & 15;

  const int rowA0 = mt * 32 + col;
  const int rowA1 = rowA0 + 16;
  const uint16_t* zrow0 = zb_new + (size_t)rowA0 * HID;
  const uint16_t* zrow1 = zb_new + (size_t)rowA1 * HID;
  const uint16_t* wcol  = WoT + (size_t)(nt * 64 + col) * HID;
  const int koffB = half * 8;

  FragF acc[2][4];
#pragma unroll
  for (int m = 0; m < 2; ++m)
#pragma unroll
    for (int s = 0; s < 4; ++s)
#pragma unroll
      for (int i = 0; i < 8; ++i) acc[m][s].f[i] = 0.0f;

#pragma unroll 2
  for (int kb = 0; kb < HID; kb += 32) {
    FragBF af[2];
    load_a_bf16(af[0], zrow0 + kb, half8);
    load_a_bf16(af[1], zrow1 + kb, half8);
    FragBF bf[4];
#pragma unroll
    for (int s = 0; s < 4; ++s) {
      const uint16_t* wc = wcol + (size_t)s * 16 * HID + kb + koffB;
      bf[s].q[0] = *(const uint4*)(wc);
      bf[s].q[1] = *(const uint4*)(wc + 16);
    }
#pragma unroll
    for (int s = 0; s < 4; ++s) {
#pragma unroll
      for (int m = 0; m < 2; ++m) {
        acc[m][s].v = __builtin_amdgcn_wmma_f32_16x16x32_bf16(
            false, af[m].v, false, bf[s].v, (short)0, acc[m][s].v, false, false);
      }
    }
  }

  float* outT = outputs + (size_t)t * BATCH * OUT_DIM;
#pragma unroll
  for (int s = 0; s < 4; ++s) {
    const int o = nt * 64 + s * 16 + col;
    const float ao = aout[o], omo = 1.0f - ao;
#pragma unroll
    for (int m = 0; m < 2; ++m) {
#pragma unroll
      for (int r = 0; r < 8; ++r) {
        const int b = mt * 32 + m * 16 + half8 + r;
        const size_t idx = (size_t)b * OUT_DIM + o;
        const float v = ao * ou[idx] + omo * acc[m][s].f[r];
        ou[idx] = v;
        outT[idx] = v;
      }
    }
  }
}

// ---------------------------------------------------------------------------
// finalize: write (zf, uf, af, ouf, ns) tail of d_out
// ---------------------------------------------------------------------------
__global__ void finalize_kernel(const uint16_t* __restrict__ zb_fin,
                                const float* __restrict__ u, const float* __restrict__ a,
                                const float* __restrict__ ou, const int* __restrict__ ns,
                                float* __restrict__ out_tail)
{
  const int stride = gridDim.x * blockDim.x;
  const int tid0 = blockIdx.x * blockDim.x + threadIdx.x;
  const int NH = BATCH * HID, NO = BATCH * OUT_DIM;
  for (int i = tid0; i < NH; i += stride) {
    out_tail[i]           = bf2f(zb_fin[i]);
    out_tail[NH + i]      = u[i];
    out_tail[2 * NH + i]  = a[i];
  }
  for (int i = tid0; i < NO; i += stride) out_tail[3 * NH + i] = ou[i];
  if (tid0 == 0) out_tail[3 * NH + NO] = (float)(*ns);
}

// ---------------------------------------------------------------------------
extern "C" void kernel_launch(void* const* d_in, const int* in_sizes, int n_in,
                              void* d_out, int out_size, void* d_ws, size_t ws_size,
                              hipStream_t stream)
{
  const float* x       = (const float*)d_in[0];
  const float* Wh      = (const float*)d_in[1];
  const float* Wo      = (const float*)d_in[2];
  const float* tau_mem = (const float*)d_in[3];
  const float* tau_adp = (const float*)d_in[4];
  const float* tau_out = (const float*)d_in[5];
  float* out = (float*)d_out;

  char* ws = (char*)d_ws;
  size_t off = 0;
  auto alloc = [&](size_t bytes) -> char* {
    char* p = ws + off;
    off += (bytes + 255) & ~(size_t)255;
    return p;
  };
  uint16_t* WhT = (uint16_t*)alloc((size_t)KTOT * HID * 2);
  uint16_t* WoT = (uint16_t*)alloc((size_t)HID * OUT_DIM * 2);
  uint16_t* zbA = (uint16_t*)alloc((size_t)BATCH * HID * 2);
  uint16_t* zbB = (uint16_t*)alloc((size_t)BATCH * HID * 2);
  float* u     = (float*)alloc((size_t)BATCH * HID * 4);
  float* a     = (float*)alloc((size_t)BATCH * HID * 4);
  float* ou    = (float*)alloc((size_t)BATCH * OUT_DIM * 4);
  float* alpha = (float*)alloc(HID * 4);
  float* rho   = (float*)alloc(HID * 4);
  float* aout  = (float*)alloc(OUT_DIM * 4);
  int*   ns    = (int*)alloc(4);
  // xb last, optional: only if workspace is big enough (decision is a pure
  // function of ws_size -> deterministic across calls)
  const size_t xb_bytes = (size_t)T_STEPS * BATCH * IN_DIM * 2;
  uint16_t* xb = nullptr;
  if (ws_size >= off + xb_bytes) xb = (uint16_t*)alloc(xb_bytes);
  (void)in_sizes; (void)n_in; (void)out_size;

  uint16_t* zb[2] = { zbA, zbB };

  init_kernel<<<512, 256, 0, stream>>>(x, Wh, Wo, tau_mem, tau_adp, tau_out,
                                       WhT, WoT, xb, zbA, zbB, u, a, ou,
                                       alpha, rho, aout, ns);

  for (int t = 0; t < T_STEPS; ++t) {
    const uint16_t* zo = zb[t & 1];
    uint16_t*       zn = zb[(t + 1) & 1];
    if (xb) {
      step_rec_kernel<true><<<32, 128, 0, stream>>>(x, xb, t, WhT, zo, zn, u, a, alpha, rho, ns);
    } else {
      step_rec_kernel<false><<<32, 128, 0, stream>>>(x, xb, t, WhT, zo, zn, u, a, alpha, rho, ns);
    }
    step_out_kernel<<<4, 128, 0, stream>>>(zn, WoT, ou, aout, out, t);
  }

  finalize_kernel<<<256, 256, 0, stream>>>(zb[T_STEPS & 1], u, a, ou, ns,
                                           out + (size_t)T_STEPS * BATCH * OUT_DIM);
}
